// LNN_81724637708829
// MI455X (gfx1250) — matrix operand order repaired and checked by reading, not compiled, not run on hardware
//
#include <hip/hip_runtime.h>
#include <hip/hip_bf16.h>

// ---------------------------------------------------------------------------
// CfC mixed-memory RNN scan for MI455X (gfx1250, wave32, WMMA bf16 16x16x32).
// B=256, T=512, IN=16, H=512, BB=128, OUT=2.
// 16 workgroups x 16 batch rows; 8 waves split the N dimension of each GEMM.
// LSTM cell state c lives in WMMA accumulator registers for all 512 steps.
// Accumulator chains start from a zero C operand; biases are folded into the
// elementwise epilogues (removes spilled bias-broadcast tiles).
// ---------------------------------------------------------------------------

#define DI __device__ __forceinline__

typedef __attribute__((ext_vector_type(16))) __bf16 bf16x16;
typedef __attribute__((ext_vector_type(8)))  __bf16 bf16x8;
typedef __attribute__((ext_vector_type(8)))  float  f32x8;

#define Bn   256
#define Tn   512
#define INn  16
#define Hn   512
#define BBn  128
#define OUTn 2

// LDS row strides (bf16 elements); stride_dwords mod 64 in {4,20} ->
// conflict-free 16-lane b128 A-tile loads; multiples of 8 -> 16B aligned.
#define CAT_S 552   // [xt(16) | zeros(16) | h_lstm(512)], 544 used
#define H_S   520   // h carry (CfC output), 512 used
#define BB_S  136   // backbone activations, 128 used

// d_ws layout (bf16 element offsets)
#define WR_OFF   0u                         // Wr       [2048][512]
#define WI_OFF   1048576u                   // Wi pad   [2048][32]
#define WBX_OFF  (WI_OFF + 65536u)          // Wbb x-part pad [128][32]
#define WBH_OFF  (WBX_OFF + 4096u)          // Wbb h-part     [128][512]
#define WF_OFF   (WBH_OFF + 65536u)         // Wf1,Wf2,Wta,Wtb each [512][128]
#define WF_SZ    65536u

#define HOFF 262144   // B*T*OUT floats
#define COFF 393216   // + B*H floats

DI float sigm_(float x) { return 1.0f / (1.0f + __expf(-x)); }
DI float tanh_(float x) { return 2.0f * sigm_(2.0f * x) - 1.0f; }   // overflow-safe

DI bf16x16 cat8(bf16x8 lo, bf16x8 hi) {
  return __builtin_shufflevector(lo, hi, 0,1,2,3,4,5,6,7,8,9,10,11,12,13,14,15);
}

DI f32x8 wmma_bf16(bf16x16 a, bf16x16 b, f32x8 c) {
  return __builtin_amdgcn_wmma_f32_16x16x32_bf16(false, a, false, b, (short)0, c,
                                                 false, false);
}

// A-matrix 16x32 tile: laneBase = lds + (lane&15)*stride + (lane>>4)*16.
DI bf16x16 loadA_off(const __bf16* laneBase, int off) {
  bf16x8 lo = *(const bf16x8*)(laneBase + off);
  bf16x8 hi = *(const bf16x8*)(laneBase + off + 8);
  return cat8(lo, hi);
}

// B-matrix 32x16 tile from row-major W[N][K]: laneBase = W + (lane&15)*K +
// (lane>>4)*16; off = n0*K + k0.
DI bf16x16 loadB_off(const __bf16* laneBase, int off) {
  bf16x8 lo = *(const bf16x8*)(laneBase + off);
  bf16x8 hi = *(const bf16x8*)(laneBase + off + 8);
  return cat8(lo, hi);
}

// ---- weight conversion / repacking kernels ----
__global__ void cvt_f32_bf16(const float* __restrict__ in, __bf16* __restrict__ out,
                             int n) {
  int i = blockIdx.x * 256 + threadIdx.x;
  if (i < n) out[i] = (__bf16)in[i];
}

// out[r][0..31] = {in[r][0..15], zeros}  (in row stride = incols)
__global__ void cvt_pad32(const float* __restrict__ in, __bf16* __restrict__ out,
                          int rows, int incols) {
  int i = blockIdx.x * 256 + threadIdx.x;
  if (i < rows * 32) {
    int r = i >> 5, k = i & 31;
    out[i] = (k < 16) ? (__bf16)in[r * incols + k] : (__bf16)0.0f;
  }
}

// out[r][0..511] = in[r][16..527]  (Wbb h-part slice)
__global__ void cvt_wbbh(const float* __restrict__ in, __bf16* __restrict__ out) {
  int i = blockIdx.x * 256 + threadIdx.x;
  if (i < BBn * Hn) {
    int r = i >> 9, k = i & 511;
    out[i] = (__bf16)in[r * (INn + Hn) + INn + k];
  }
}

__global__ __launch_bounds__(256)
void cfc_scan_kernel(const float* __restrict__ x,
                     const float* __restrict__ bi,  const float* __restrict__ bbb,
                     const float* __restrict__ bf1, const float* __restrict__ bf2,
                     const float* __restrict__ bta, const float* __restrict__ btb,
                     const float* __restrict__ Wh,  const float* __restrict__ bh,
                     const __bf16* __restrict__ wr, const __bf16* __restrict__ wi,
                     const __bf16* __restrict__ wbx, const __bf16* __restrict__ wbh,
                     const __bf16* __restrict__ wf,
                     float* __restrict__ out) {
  __shared__ __align__(16) __bf16 s_cat[16 * CAT_S];   // [xt | 0 | h_lstm]
  __shared__ __align__(16) __bf16 s_h[16 * H_S];       // CfC h carry
  __shared__ __align__(16) __bf16 s_bb[16 * BB_S];     // backbone activations
  __shared__ float s_out[32];                          // head accumulator [16][2]

  const int tid  = threadIdx.x;
  const int wv   = tid >> 5;
  const int lane = tid & 31;
  const int nl   = lane & 15;
  const int hf   = lane >> 4;          // C/D tiles: row m = vgpr + 8*hf, col = nl
  const int b0   = blockIdx.x * 16;

  // per-lane base pointers (tile offsets become instruction immediates)
  const __bf16* wrB  = wr  + nl * Hn  + hf * 16;
  const __bf16* wiB  = wi  + nl * 32  + hf * 16;
  const __bf16* wbxB = wbx + nl * 32  + hf * 16;
  const __bf16* wbhB = wbh + nl * Hn  + hf * 16;
  const __bf16* wfB  = wf  + nl * BBn + hf * 16;
  const __bf16* catA = s_cat + nl * CAT_S + hf * 16;
  const __bf16* hA   = s_h   + nl * H_S   + hf * 16;
  const __bf16* bbA  = s_bb  + nl * BB_S  + hf * 16;

  // ---- init: zero cat (incl. the 16-col zero gap) and h carry ----
  for (int i = tid; i < 16 * CAT_S; i += 256) s_cat[i] = (__bf16)0.0f;
  for (int i = tid; i < 16 * H_S; i += 256) s_h[i] = (__bf16)0.0f;

  f32x8 c_[4];
  #pragma unroll
  for (int nt = 0; nt < 4; ++nt)
    #pragma unroll
    for (int e = 0; e < 8; ++e) c_[nt][e] = 0.0f;

  // scalar biases, folded into the elementwise epilogues
  float biasA[16], biasT[4], biasF1[4], biasF2[4], wh0[4], wh1[4];
  #pragma unroll
  for (int g = 0; g < 4; ++g)
    #pragma unroll
    for (int nt = 0; nt < 4; ++nt)
      biasA[g * 4 + nt] = bi[g * Hn + wv * 64 + nt * 16 + nl];
  #pragma unroll
  for (int nt = 0; nt < 4; ++nt) {
    int col = wv * 64 + nt * 16 + nl;
    biasT[nt]  = bta[col] + btb[col];   // t_interp biases pre-summed
    biasF1[nt] = bf1[col];
    biasF2[nt] = bf2[col];
    wh0[nt] = Wh[col];
    wh1[nt] = Wh[Hn + col];
  }
  const float biasB = bbb[wv * 16 + nl];
  const float bhv = (tid < 32) ? bh[tid & 1] : 0.0f;

  f32x8 zf;
  #pragma unroll
  for (int e = 0; e < 8; ++e) zf[e] = 0.0f;

  for (int t = 0; t < Tn; ++t) {
    // Defeat LICM without breaking global-addrspace inference: an opaque
    // uniform zero is re-produced each iteration and folded into the bases.
    int zofs = 0;
    asm volatile("" : "+s"(zofs));
    const __bf16* wrBt  = wrB  + zofs;
    const __bf16* wiBt  = wiB  + zofs;
    const __bf16* wbxBt = wbxB + zofs;
    const __bf16* wbhBt = wbhB + zofs;
    const __bf16* wfBt  = wfB  + zofs;

    // ---- stage xt into cat cols 0..15; init head accumulator ----
    {
      int r = tid >> 4, cx = tid & 15;
      s_cat[r * CAT_S + cx] =
          (__bf16)__builtin_nontemporal_load(&x[((long)(b0 + r) * Tn + t) * INn + cx]);
    }
    if (tid < 32) s_out[tid] = bhv;
    __syncthreads();   // xt + h carry + head acc ready

    // =============== Phase A: z = xt@Wi.T + h@Wr.T (+ bi later) ===============
    bf16x16 ax = loadA_off(catA, 0);   // xt tile (cols 16..31 are zeros)

    f32x8 acc[8];
    // pass 0: gates i (g=0), ig (g=1); chains start from zero C
    #pragma unroll
    for (int g = 0; g < 2; ++g)
      #pragma unroll
      for (int nt = 0; nt < 4; ++nt)
        acc[g * 4 + nt] =
            wmma_bf16(ax, loadB_off(wiBt, (g * Hn + wv * 64 + nt * 16) * 32), zf);
    #pragma unroll 2
    for (int ki = 0; ki < 16; ++ki) {
      bf16x16 a = loadA_off(hA, ki * 32);
      #pragma unroll
      for (int g = 0; g < 2; ++g)
        #pragma unroll
        for (int nt = 0; nt < 4; ++nt)
          acc[g * 4 + nt] = wmma_bf16(
              a, loadB_off(wrBt, (g * Hn + wv * 64 + nt * 16) * Hn + ki * 32),
              acc[g * 4 + nt]);
    }
    f32x8 tin[4];   // tanh(i+bi)*sigmoid(ig+big)
    #pragma unroll
    for (int nt = 0; nt < 4; ++nt)
      #pragma unroll
      for (int e = 0; e < 8; ++e)
        tin[nt][e] = tanh_(acc[nt][e] + biasA[nt]) *
                     sigm_(acc[4 + nt][e] + biasA[4 + nt]);

    // pass 1: gates fg (g=2), og (g=3)
    #pragma unroll
    for (int g = 0; g < 2; ++g)
      #pragma unroll
      for (int nt = 0; nt < 4; ++nt)
        acc[g * 4 + nt] = wmma_bf16(
            ax, loadB_off(wiBt, ((g + 2) * Hn + wv * 64 + nt * 16) * 32), zf);
    #pragma unroll 2
    for (int ki = 0; ki < 16; ++ki) {
      bf16x16 a = loadA_off(hA, ki * 32);
      #pragma unroll
      for (int g = 0; g < 2; ++g)
        #pragma unroll
        for (int nt = 0; nt < 4; ++nt)
          acc[g * 4 + nt] = wmma_bf16(
              a, loadB_off(wrBt, ((g + 2) * Hn + wv * 64 + nt * 16) * Hn + ki * 32),
              acc[g * 4 + nt]);
    }
    // LSTM cell update; h_lstm -> cat cols 32..543 (bf16)
    #pragma unroll
    for (int nt = 0; nt < 4; ++nt)
      #pragma unroll
      for (int e = 0; e < 8; ++e) {
        float cf = c_[nt][e] * sigm_(acc[nt][e] + biasA[8 + nt] + 1.0f) + tin[nt][e];
        c_[nt][e] = cf;
        float hl = tanh_(cf) * sigm_(acc[4 + nt][e] + biasA[12 + nt]);
        s_cat[(e + 8 * hf) * CAT_S + 32 + wv * 64 + nt * 16 + nl] = (__bf16)hl;
      }
    __syncthreads();   // cat complete; h-carry reads done

    // =============== Phase B: bb = lecun_tanh(cat @ Wbb.T + bbb) ===============
    f32x8 bb0 = wmma_bf16(ax, loadB_off(wbxBt, (wv * 16) * 32), zf);
    f32x8 bb1 = zf;
    #pragma unroll 2
    for (int ki = 0; ki < 16; ki += 2) {   // two interleaved accumulator chains
      bf16x16 a0 = loadA_off(catA, 32 + ki * 32);
      bf16x16 a1 = loadA_off(catA, 32 + (ki + 1) * 32);
      bf16x16 b0 = loadB_off(wbhBt, (wv * 16) * Hn + ki * 32);
      bf16x16 b1 = loadB_off(wbhBt, (wv * 16) * Hn + (ki + 1) * 32);
      bb0 = wmma_bf16(a0, b0, bb0);
      bb1 = wmma_bf16(a1, b1, bb1);
    }
    #pragma unroll
    for (int e = 0; e < 8; ++e)
      s_bb[(e + 8 * hf) * BB_S + wv * 16 + nl] =
          (__bf16)(1.7159f * tanh_(0.666f * (bb0[e] + bb1[e] + biasB)));
    __syncthreads();   // bb ready

    // ====== Phase C: t_interp first (ta/tb), then ff1/ff2 + interpolation ======
    f32x8 acc2[8];
    // pass 0: ta (mat 2), tb (mat 3); ki = 0 starts chains from zero C
    {
      bf16x16 a = loadA_off(bbA, 0);
      #pragma unroll
      for (int mt = 0; mt < 2; ++mt)
        #pragma unroll
        for (int nt = 0; nt < 4; ++nt)
          acc2[mt * 4 + nt] = wmma_bf16(
              a, loadB_off(wfBt, (mt + 2) * (int)WF_SZ + (wv * 64 + nt * 16) * BBn),
              zf);
    }
    #pragma unroll
    for (int ki = 1; ki < 4; ++ki) {
      bf16x16 a = loadA_off(bbA, ki * 32);
      #pragma unroll
      for (int mt = 0; mt < 2; ++mt)
        #pragma unroll
        for (int nt = 0; nt < 4; ++nt)
          acc2[mt * 4 + nt] = wmma_bf16(
              a,
              loadB_off(wfBt,
                        (mt + 2) * (int)WF_SZ + (wv * 64 + nt * 16) * BBn + ki * 32),
              acc2[mt * 4 + nt]);
    }
    f32x8 tt[4];   // sigmoid(ta + tb + bta + btb)
    #pragma unroll
    for (int nt = 0; nt < 4; ++nt)
      #pragma unroll
      for (int e = 0; e < 8; ++e)
        tt[nt][e] = sigm_(acc2[nt][e] + acc2[4 + nt][e] + biasT[nt]);

    // pass 1: ff1 (mat 0), ff2 (mat 1)
    {
      bf16x16 a = loadA_off(bbA, 0);
      #pragma unroll
      for (int mt = 0; mt < 2; ++mt)
        #pragma unroll
        for (int nt = 0; nt < 4; ++nt)
          acc2[mt * 4 + nt] = wmma_bf16(
              a, loadB_off(wfBt, mt * (int)WF_SZ + (wv * 64 + nt * 16) * BBn), zf);
    }
    #pragma unroll
    for (int ki = 1; ki < 4; ++ki) {
      bf16x16 a = loadA_off(bbA, ki * 32);
      #pragma unroll
      for (int mt = 0; mt < 2; ++mt)
        #pragma unroll
        for (int nt = 0; nt < 4; ++nt)
          acc2[mt * 4 + nt] = wmma_bf16(
              a,
              loadB_off(wfBt, mt * (int)WF_SZ + (wv * 64 + nt * 16) * BBn + ki * 32),
              acc2[mt * 4 + nt]);
    }
    // h_new = tanh(ff1+b1) + t*(tanh(ff2+b2)-tanh(ff1+b1)); h carry + head
    float po0[8], po1[8];
    #pragma unroll
    for (int e = 0; e < 8; ++e) { po0[e] = 0.0f; po1[e] = 0.0f; }
    #pragma unroll
    for (int nt = 0; nt < 4; ++nt)
      #pragma unroll
      for (int e = 0; e < 8; ++e) {
        float t1 = tanh_(acc2[nt][e] + biasF1[nt]);
        float t2 = tanh_(acc2[4 + nt][e] + biasF2[nt]);
        float hn = t1 + tt[nt][e] * (t2 - t1);
        s_h[(e + 8 * hf) * H_S + wv * 64 + nt * 16 + nl] = (__bf16)hn;
        po0[e] += hn * wh0[nt];
        po1[e] += hn * wh1[nt];
      }
    // butterfly-reduce over the 16 lanes sharing a row group
    #pragma unroll
    for (int e = 0; e < 8; ++e) {
      float a0 = po0[e], a1 = po1[e];
      #pragma unroll
      for (int mk = 1; mk <= 8; mk <<= 1) {
        a0 += __shfl_xor(a0, mk, 32);
        a1 += __shfl_xor(a1, mk, 32);
      }
      po0[e] = a0; po1[e] = a1;
    }
    if (nl == 0) {
      #pragma unroll
      for (int e = 0; e < 8; ++e) {
        atomicAdd(&s_out[(e + 8 * hf) * 2 + 0], po0[e]);
        atomicAdd(&s_out[(e + 8 * hf) * 2 + 1], po1[e]);
      }
    }
    __syncthreads();   // head + h carry complete

    if (tid < 32)
      out[((long)(b0 + (tid >> 1)) * Tn + t) * OUTn + (tid & 1)] = s_out[tid];
  }

  __syncthreads();
  // ---- h_last ----
  for (int i = tid; i < 16 * Hn; i += 256) {
    int m = i >> 9, n = i & 511;
    out[HOFF + (long)(b0 + m) * Hn + n] = (float)s_h[m * H_S + n];
  }
  // ---- c_last (from registers) ----
  #pragma unroll
  for (int nt = 0; nt < 4; ++nt)
    #pragma unroll
    for (int e = 0; e < 8; ++e)
      out[COFF + (long)(b0 + e + 8 * hf) * Hn + wv * 64 + nt * 16 + nl] = c_[nt][e];
}

extern "C" void kernel_launch(void* const* d_in, const int* in_sizes, int n_in,
                              void* d_out, int out_size, void* d_ws, size_t ws_size,
                              hipStream_t stream) {
  (void)in_sizes; (void)n_in; (void)out_size; (void)ws_size;
  const float* x   = (const float*)d_in[0];
  const float* Wi  = (const float*)d_in[1];
  const float* bi  = (const float*)d_in[2];
  const float* Wr  = (const float*)d_in[3];
  const float* Wbb = (const float*)d_in[4];
  const float* bbb = (const float*)d_in[5];
  const float* Wf1 = (const float*)d_in[6];
  const float* bf1 = (const float*)d_in[7];
  const float* Wf2 = (const float*)d_in[8];
  const float* bf2 = (const float*)d_in[9];
  const float* Wta = (const float*)d_in[10];
  const float* bta = (const float*)d_in[11];
  const float* Wtb = (const float*)d_in[12];
  const float* btb = (const float*)d_in[13];
  const float* Wh  = (const float*)d_in[14];
  const float* bh  = (const float*)d_in[15];
  float* out = (float*)d_out;

  __bf16* ws  = (__bf16*)d_ws;   // ~2.9 MB of repacked bf16 weights
  __bf16* wr  = ws + WR_OFF;
  __bf16* wi  = ws + WI_OFF;
  __bf16* wbx = ws + WBX_OFF;
  __bf16* wbh = ws + WBH_OFF;
  __bf16* wf  = ws + WF_OFF;     // Wf1|Wf2|Wta|Wtb contiguous

  auto cvt = [&](const float* src, __bf16* dst, int n) {
    cvt_f32_bf16<<<(n + 255) / 256, 256, 0, stream>>>(src, dst, n);
  };
  cvt(Wr, wr, 4 * Hn * Hn);                                   // 2048*512
  cvt_pad32<<<(4 * Hn * 32 + 255) / 256, 256, 0, stream>>>(Wi, wi, 4 * Hn, INn);
  cvt_pad32<<<(BBn * 32 + 255) / 256, 256, 0, stream>>>(Wbb, wbx, BBn, INn + Hn);
  cvt_wbbh<<<(BBn * Hn + 255) / 256, 256, 0, stream>>>(Wbb, wbh);
  cvt(Wf1, wf + 0 * WF_SZ, Hn * BBn);
  cvt(Wf2, wf + 1 * WF_SZ, Hn * BBn);
  cvt(Wta, wf + 2 * WF_SZ, Hn * BBn);
  cvt(Wtb, wf + 3 * WF_SZ, Hn * BBn);

  cfc_scan_kernel<<<Bn / 16, 256, 0, stream>>>(
      x, bi, bbb, bf1, bf2, bta, btb, Wh, bh,
      wr, wi, wbx, wbh, wf, out);
}
